// MyBertSelfAttention9_42245298324248
// MI455X (gfx1250) — matrix operand order repaired and checked by reading, not compiled, not run on hardware
//
#include <hip/hip_runtime.h>

// ---------------------------------------------------------------------------
// Problem constants
// ---------------------------------------------------------------------------
#define BB   16          // batch
#define SS   512         // sequence
#define HID  1024
#define NH   16          // heads
#define HD   64          // head dim
#define BLK  256         // FiLM block
#define MTOK (BB*SS)     // 8192 tokens

typedef __bf16 bf16_t;
typedef __attribute__((ext_vector_type(16))) __bf16 v16bf;
typedef __attribute__((ext_vector_type(8)))  __bf16 v8bf;
typedef __attribute__((ext_vector_type(4)))  __bf16 v4bf;
typedef __attribute__((ext_vector_type(8)))  float  v8f;

static __device__ inline v8f wmma_bf16(v16bf a, v16bf b, v8f c) {
  // D = A(16x32 bf16) * B(32x16 bf16) + C(16x16 f32)
  return __builtin_amdgcn_wmma_f32_16x16x32_bf16(
      /*neg_a=*/false, a, /*neg_b=*/false, b,
      /*c_mod=*/(short)0, c, /*reuse_a=*/false, /*reuse_b=*/false);
}

static __device__ inline v8f vzero8f() {
  v8f z = {0.f,0.f,0.f,0.f,0.f,0.f,0.f,0.f};
  return z;
}

// Build 16x32 bf16 A-fragment for one lane.
// A layout (16-bit, 16x32), contiguous K order matching B: lanes 0-15 hold K=0..15;
// lanes 16-31 hold K=16..31. Two contiguous 16B chunks per lane.
template <typename PTR>
static __device__ inline v16bf make_a_frag(PTR rowPtr, int k0, int laneHi) {
  const int base = k0 + (laneHi ? 16 : 0);
  v8bf lo = *(const v8bf*)(rowPtr + base);
  v8bf hi = *(const v8bf*)(rowPtr + base + 8);
  return __builtin_shufflevector(lo, hi,
      0,1,2,3,4,5,6,7, 8,9,10,11,12,13,14,15);
}

// ---------------------------------------------------------------------------
// Kernel 1: f32 -> bf16 convert (vectorized x4)
// ---------------------------------------------------------------------------
__global__ __launch_bounds__(256) void cvt_bf16_kernel(const float* __restrict__ in,
                                                       bf16_t* __restrict__ out, int n4) {
  int i = blockIdx.x * blockDim.x + threadIdx.x;
  if (i < n4) {
    float4 f = ((const float4*)in)[i];
    v4bf o = { (__bf16)f.x, (__bf16)f.y, (__bf16)f.z, (__bf16)f.w };
    ((v4bf*)out)[i] = o;
  }
}

// ---------------------------------------------------------------------------
// Kernel 2: FiLM gamma/beta = task_emb @ W^T + b     [B, BLK] each
// ---------------------------------------------------------------------------
__global__ __launch_bounds__(256) void film_kernel(const float* __restrict__ te,
                                                   const float* __restrict__ Wg,
                                                   const float* __restrict__ bg,
                                                   const float* __restrict__ Wb,
                                                   const float* __restrict__ bb,
                                                   float* __restrict__ gamma,
                                                   float* __restrict__ beta) {
  int t = blockIdx.x * blockDim.x + threadIdx.x;   // 2 * B * BLK = 8192
  int isBeta = t >> 12;
  int b = (t >> 8) & (BB - 1);
  int c = t & (BLK - 1);
  const float* w = (isBeta ? Wb : Wg) + (size_t)c * HID;
  const float* x = te + (size_t)b * HID;
  float acc = 0.f;
  #pragma unroll 4
  for (int k = 0; k < HID; ++k) acc = fmaf(x[k], w[k], acc);
  acc += isBeta ? bb[c] : bg[c];
  (isBeta ? beta : gamma)[b * BLK + c] = acc;
}

// ---------------------------------------------------------------------------
// Kernel 3: fused QKV GEMM with async-to-LDS B staging.
// Workgroup = 8 waves, tile = 128(M) x 64(N), all three weight matrices.
// B tiles (3 x 32K x 64N bf16 = 12 KB/step) are fetched once per workgroup via
// global_load_async_to_lds_b128 (ASYNCcnt), double-buffered in LDS.
// Each wave owns a 16x64 strip; A fragment reused by 12 WMMAs per k-step.
// Outputs: q,k as [b,h,s,d] bf16 (q scaled 1/8);  v transposed [b,h,d,s] bf16.
// ---------------------------------------------------------------------------
__global__ __launch_bounds__(256) void qkv_kernel(const bf16_t* __restrict__ hsb,
                                                  const bf16_t* __restrict__ wq,
                                                  const bf16_t* __restrict__ wk,
                                                  const bf16_t* __restrict__ wv,
                                                  const float* __restrict__ bq,
                                                  const float* __restrict__ bk,
                                                  const float* __restrict__ bv,
                                                  bf16_t* __restrict__ qo,
                                                  bf16_t* __restrict__ ko,
                                                  bf16_t* __restrict__ vto) {
  // [buf][weight][ n(64) rows x k(32) ] bf16 : 2*3*4KB = 24 KB
  __shared__ __align__(32) bf16_t sB[2][3][64 * 32];

  const int tid    = threadIdx.x;
  const int waveId = tid >> 5;          // 0..7
  const int lane   = tid & 31;
  const int laneHi = lane >> 4;
  const int laneLo = lane & 15;
  const int mg = blockIdx.x & 63;       // 64 M-groups of 128 rows
  const int ng = blockIdx.x >> 6;       // 16 N-groups of 64 cols
  const int m0 = (mg << 7) + (waveId << 4);
  const int n0 = ng << 6;

  // Cooperative async-loader mapping: 256 threads x 16B cover one 64x32 tile.
  const int lrow   = tid >> 2;          // 0..63 : n within group
  const int lchunk = tid & 3;           // 16B chunk within 64B row
  const bf16_t* Wsrc[3] = { wq, wk, wv };
  const size_t gRowOff = (size_t)(n0 + lrow) * HID + lchunk * 8;

  auto issue_async = [&](int buf, int k0) {
    #pragma unroll
    for (int w = 0; w < 3; ++w) {
      unsigned long long ga =
          (unsigned long long)(const void*)(Wsrc[w] + gRowOff + k0);
      // LDS aperture is 4GB-aligned: low 32 bits of the flat address are the
      // LDS byte offset the async-DMA VDST operand expects.
      unsigned la = (unsigned)(unsigned long long)(const void*)
                    &sB[buf][w][lrow * 32 + lchunk * 8];
      asm volatile("global_load_async_to_lds_b128 %0, %1, off"
                   :: "v"(la), "v"(ga) : "memory");
    }
  };

  v8f accq[4], acck[4], accv[4];
  #pragma unroll
  for (int t = 0; t < 4; ++t) { accq[t] = vzero8f(); acck[t] = vzero8f(); accv[t] = vzero8f(); }

  const bf16_t* arow = hsb + (size_t)(m0 + laneLo) * HID;

  issue_async(0, 0);
  for (int k0 = 0; k0 < HID; k0 += 32) {
    const int buf = (k0 >> 5) & 1;
    asm volatile("s_wait_asynccnt 0x0" ::: "memory");   // this wave's copies done
    __syncthreads();                                    // whole tile visible
    if (k0 + 32 < HID) issue_async(buf ^ 1, k0 + 32);   // prefetch next step

    v16bf a = make_a_frag(arow, k0, laneHi);
    #pragma unroll
    for (int t = 0; t < 4; ++t) {
      // B frag: lane holds col N = t*16+laneLo, 16 contiguous K per lane half.
      const int bi = (t * 16 + laneLo) * 32 + 16 * laneHi;
      v16bf fq = *(const v16bf*)&sB[buf][0][bi];
      v16bf fk = *(const v16bf*)&sB[buf][1][bi];
      v16bf fv = *(const v16bf*)&sB[buf][2][bi];
      accq[t] = wmma_bf16(a, fq, accq[t]);
      acck[t] = wmma_bf16(a, fk, acck[t]);
      accv[t] = wmma_bf16(a, fv, accv[t]);
    }
    __syncthreads();                                    // done reading buf
  }

  // Epilogue: bias add, Q scaled by 1/sqrt(64), head-split layouts.
  #pragma unroll
  for (int t = 0; t < 4; ++t) {
    const int n  = n0 + t*16 + laneLo;
    const float vbq = bq[n], vbk = bk[n], vbv = bv[n];
    const int h = n >> 6, dd = n & (HD - 1);
    #pragma unroll
    for (int r = 0; r < 8; ++r) {
      const int m = m0 + r + 8*laneHi;        // token index
      const int b = m >> 9, s = m & (SS - 1);
      const size_t bh = (size_t)(b * NH + h);
      const size_t qidx = (bh * SS + s) * HD + dd;
      qo[qidx] = (__bf16)((accq[t][r] + vbq) * 0.125f);
      ko[qidx] = (__bf16)(acck[t][r] + vbk);
      const size_t vidx = (bh * HD + dd) * SS + s;
      vto[vidx] = (__bf16)(accv[t][r] + vbv);
    }
  }
}

// ---------------------------------------------------------------------------
// Kernel 4: attention. 1 wave per (b, h, 16-row q-tile).
// scores (16x512) via WMMA -> LDS f32 (+FiLM block-diag bias), softmax,
// probs bf16 -> LDS, ctx = P @ V^T via WMMA, f32 out.
// K/V fragments are software-pipelined 1 step ahead into distinct registers.
// ---------------------------------------------------------------------------
__global__ __launch_bounds__(32) void attn_kernel(const bf16_t* __restrict__ qbh,
                                                  const bf16_t* __restrict__ kbh,
                                                  const bf16_t* __restrict__ vT,
                                                  const float* __restrict__ gamma,
                                                  const float* __restrict__ beta,
                                                  const float* __restrict__ rwm,
                                                  float* __restrict__ out) {
  __shared__ float  sS[16 * SS];       // 32 KB
  __shared__ __align__(32) bf16_t sP[16 * SS];   // 16 KB

  const int blk = blockIdx.x;          // B*NH*32 = 8192
  const int qt  = blk & 31;
  const int h   = (blk >> 5) & (NH - 1);
  const int b   = blk >> 9;
  const int lane   = threadIdx.x;
  const int laneHi = lane >> 4;
  const int laneLo = lane & 15;
  const size_t bh  = (size_t)(b * NH + h);

  // Q fragments for this tile stay in registers (K = d = 64 -> two 16x32 frags)
  const bf16_t* qrow = qbh + (bh * SS + qt*16 + laneLo) * HD;
  const v16bf a0 = make_a_frag(qrow, 0,  laneHi);
  const v16bf a1 = make_a_frag(qrow, 32, laneHi);

  const int qblk = qt >> 4;            // which 256-block the q rows live in
  const bf16_t* kbase = kbh + bh * SS * HD;

  // ---- scores: software-pipelined over 32 key tiles ----
  v16bf c0 = *(const v16bf*)(kbase + (size_t)laneLo * HD + 16*laneHi);
  v16bf c1 = *(const v16bf*)(kbase + (size_t)laneLo * HD + 32 + 16*laneHi);
  for (int jt = 0; jt < 32; ++jt) {
    v16bf nf0 = c0, nf1 = c1;
    if (jt + 1 < 32) {
      const bf16_t* kr = kbase + (size_t)((jt+1)*16 + laneLo) * HD;
      nf0 = *(const v16bf*)(kr + 16*laneHi);
      nf1 = *(const v16bf*)(kr + 32 + 16*laneHi);
    }
    v8f acc = vzero8f();
    acc = wmma_bf16(a0, c0, acc);
    acc = wmma_bf16(a1, c1, acc);

    const int j = jt*16 + laneLo;      // key index for this lane
    if ((jt >> 4) == qblk) {           // uniform per tile: no divergence
      const int c = j & (BLK - 1);
      const float g  = gamma[b * BLK + c];
      const float be = beta[b * BLK + c];
      #pragma unroll
      for (int r = 0; r < 8; ++r) {
        const int i = qt*16 + r + 8*laneHi;          // query row (global in S)
        acc[r] += fmaf(g, rwm[i * BLK + c], be);
      }
    }
    #pragma unroll
    for (int r = 0; r < 8; ++r)
      sS[(r + 8*laneHi) * SS + j] = acc[r];
    c0 = nf0; c1 = nf1;
  }
  __syncthreads();

  // ---- softmax over 512 keys: two lanes per row ----
  {
    float* rp = sS + laneLo * SS + laneHi * 256;
    float m = -__builtin_inff();
    for (int c = 0; c < 256; ++c) m = fmaxf(m, rp[c]);
    m = fmaxf(m, __shfl_xor(m, 16));
    const float kln2 = 1.4426950408889634f;
    float ssum = 0.f;
    for (int c = 0; c < 256; ++c) {
      float e = __builtin_amdgcn_exp2f((rp[c] - m) * kln2);
      rp[c] = e;
      ssum += e;
    }
    ssum += __shfl_xor(ssum, 16);
    const float inv = 1.0f / ssum;
    bf16_t* pp = sP + laneLo * SS + laneHi * 256;
    for (int c = 0; c < 256; ++c) pp[c] = (__bf16)(rp[c] * inv);
  }
  __syncthreads();

  // ---- ctx = P(16x512) @ V^T(512x64): 4 N-tiles, 16 K-steps, pipelined ----
  v8f ctx[4];
  #pragma unroll
  for (int t = 0; t < 4; ++t) ctx[t] = vzero8f();

  const bf16_t* vbase = vT + bh * HD * SS;
  v16bf bcur[4];
  #pragma unroll
  for (int t = 0; t < 4; ++t)
    bcur[t] = *(const v16bf*)(vbase + (size_t)(t*16 + laneLo) * SS + 16*laneHi);

  for (int k0 = 0; k0 < SS; k0 += 32) {
    v16bf bnext[4];
    #pragma unroll
    for (int t = 0; t < 4; ++t) bnext[t] = bcur[t];
    if (k0 + 32 < SS) {
      #pragma unroll
      for (int t = 0; t < 4; ++t)
        bnext[t] = *(const v16bf*)(vbase + (size_t)(t*16 + laneLo) * SS
                                   + (k0 + 32) + 16*laneHi);
    }
    v16bf a = make_a_frag(sP + laneLo * SS, k0, laneHi);
    #pragma unroll
    for (int t = 0; t < 4; ++t) ctx[t] = wmma_bf16(a, bcur[t], ctx[t]);
    #pragma unroll
    for (int t = 0; t < 4; ++t) bcur[t] = bnext[t];
  }

  #pragma unroll
  for (int t = 0; t < 4; ++t) {
    const int dd = t*16 + laneLo;
    #pragma unroll
    for (int r = 0; r < 8; ++r) {
      const int s = qt*16 + r + 8*laneHi;
      out[((size_t)(b * SS + s)) * HID + h * HD + dd] = ctx[t][r];
    }
  }
}

// ---------------------------------------------------------------------------
// Launch
// ---------------------------------------------------------------------------
extern "C" void kernel_launch(void* const* d_in, const int* in_sizes, int n_in,
                              void* d_out, int out_size, void* d_ws, size_t ws_size,
                              hipStream_t stream) {
  const float* hs  = (const float*)d_in[0];
  const float* te  = (const float*)d_in[1];
  const float* Wq  = (const float*)d_in[2];
  const float* bq  = (const float*)d_in[3];
  const float* Wk  = (const float*)d_in[4];
  const float* bk  = (const float*)d_in[5];
  const float* Wv  = (const float*)d_in[6];
  const float* bv  = (const float*)d_in[7];
  const float* Wg  = (const float*)d_in[8];
  const float* bg  = (const float*)d_in[9];
  const float* Wb  = (const float*)d_in[10];
  const float* bb  = (const float*)d_in[11];
  const float* rwm = (const float*)d_in[12];
  float* out = (float*)d_out;

  char* ws = (char*)d_ws;
  size_t off = 0;
  bf16_t* hsb = (bf16_t*)(ws + off); off += (size_t)MTOK * HID * 2;   // 16 MiB
  bf16_t* wqb = (bf16_t*)(ws + off); off += (size_t)HID * HID * 2;    //  2 MiB
  bf16_t* wkb = (bf16_t*)(ws + off); off += (size_t)HID * HID * 2;
  bf16_t* wvb = (bf16_t*)(ws + off); off += (size_t)HID * HID * 2;
  bf16_t* qbh = (bf16_t*)(ws + off); off += (size_t)MTOK * HID * 2;   // [b,h,s,d]
  bf16_t* kbh = (bf16_t*)(ws + off); off += (size_t)MTOK * HID * 2;   // [b,h,s,d]
  bf16_t* vtb = (bf16_t*)(ws + off); off += (size_t)MTOK * HID * 2;   // [b,h,d,s]
  float* gamma = (float*)(ws + off); off += (size_t)BB * BLK * 4;
  float* beta  = (float*)(ws + off); off += (size_t)BB * BLK * 4;

  // 1) convert activations + weights to bf16
  {
    int n4 = MTOK * HID / 4;
    cvt_bf16_kernel<<<(n4 + 255) / 256, 256, 0, stream>>>(hs, hsb, n4);
    int w4 = HID * HID / 4;
    cvt_bf16_kernel<<<(w4 + 255) / 256, 256, 0, stream>>>(Wq, wqb, w4);
    cvt_bf16_kernel<<<(w4 + 255) / 256, 256, 0, stream>>>(Wk, wkb, w4);
    cvt_bf16_kernel<<<(w4 + 255) / 256, 256, 0, stream>>>(Wv, wvb, w4);
  }

  // 2) FiLM gamma/beta
  film_kernel<<<(2 * BB * BLK) / 256, 256, 0, stream>>>(te, Wg, bg, Wb, bb, gamma, beta);

  // 3) fused QKV GEMM: 64 M-groups x 16 N-groups, 8 waves/block
  qkv_kernel<<<1024, 256, 0, stream>>>(hsb, wqb, wkb, wvb, bq, bk, bv, qbh, kbh, vtb);

  // 4) attention: one wave per (b, h, q-tile)
  attn_kernel<<<BB * NH * 32, 32, 0, stream>>>(qbh, kbh, vtb, gamma, beta, rwm, out);
}
